// ManualAttention_46248207843356
// MI455X (gfx1250) — matrix operand order repaired and checked by reading, not compiled, not run on hardware
//
#include <hip/hip_runtime.h>
#include <hip/hip_bf16.h>

// ---------------------------------------------------------------------------
// Types for CDNA5 WMMA (wave32): v_wmma_f32_16x16x32_bf16
// ---------------------------------------------------------------------------
typedef __bf16 bf16_t;
typedef __attribute__((ext_vector_type(16))) __bf16 v16bf;
typedef __attribute__((ext_vector_type(8)))  __bf16 v8bf;
typedef __attribute__((ext_vector_type(4)))  __bf16 v4bf;
typedef __attribute__((ext_vector_type(8)))  float  v8f;
typedef __attribute__((ext_vector_type(4)))  unsigned int v4u;
typedef __attribute__((ext_vector_type(8)))  int    v8i;
typedef __attribute__((ext_vector_type(4)))  int    v4i;

union FragU { v16bf v; v8bf h[2]; };

// Problem constants
#define BATCH 8
#define SEQ   1024
#define DIM   768
#define HEADS 12
#define HDIM  64
#define ROWS  (BATCH * SEQ)       // 8192
#define QKV_N (3 * DIM)           // 2304
#define ATTN_SCALE 0.125f         // 64^-0.5
#define PAD_LD 72                 // 64 + 8 elem pad -> 36-dword LDS row stride
                                  // (36*c mod 64 distinct for c=0..15: conflict-free)

// ---------------------------------------------------------------------------
// Fragment loaders matching ISA 7.12.2 VGPR layouts (wave32).
// A (16x32 MxK) row-major:  lane=half*16+m; elems 0..7 -> K=k0+half*8+0..7,
//                           elems 8..15 -> K=k0+16+half*8+0..7
// B (32x16 KxN) where B[k,n]=Wt[n*ld+k]:  lane=half*16+c (col n0+c);
//                           elems 0..7 -> K=k0+half*8+0..7, elems 8..15 -> K=k0+16+half*8+0..7
// Each loader = two 16-byte loads per lane.
// ---------------------------------------------------------------------------
__device__ __forceinline__ v16bf load_a_frag(const bf16_t* base, int ld,
                                             int row0, int k0, int lane) {
  int half = (lane >> 4) & 1;
  int m    = lane & 15;
  const bf16_t* p = base + (size_t)(row0 + m) * ld + k0;
  FragU f;
  f.h[0] = *(const v8bf*)(p + half * 8);
  f.h[1] = *(const v8bf*)(p + 16 + half * 8);
  return f.v;
}

__device__ __forceinline__ v16bf load_bt_frag(const bf16_t* base, int ld,
                                              int n0, int k0, int lane) {
  int half = (lane >> 4) & 1;
  int c    = lane & 15;
  const bf16_t* p = base + (size_t)(n0 + c) * ld + k0 + half * 8;
  FragU f;
  f.h[0] = *(const v8bf*)(p);
  f.h[1] = *(const v8bf*)(p + 16);
  return f.v;
}

__device__ __forceinline__ v8f wmma_bf16(v16bf a, v16bf b, v8f c) {
  return __builtin_amdgcn_wmma_f32_16x16x32_bf16(false, a, false, b,
                                                 (short)0, c, false, false);
}

// ---------------------------------------------------------------------------
// Ping-pong double-buffered 32x64 per-wave GEMM tile.
// Tail-peeled (no wrap-select -> constant +64 prefetch distance) and fully
// unrolled (K is a compile-time 768 after inlining): all loads become
// base + immediate-offset, removing the per-iteration 64-bit address
// arithmetic and its WMMA->VALU hazard v_nops seen in round 3.
// ---------------------------------------------------------------------------
__device__ __forceinline__ void gemm_tile_32x64(const bf16_t* A, int lda,
                                                const bf16_t* Wt, int ldw,
                                                int row0, int n0, int K,
                                                v8f acc[8], int lane) {
  v16bf a0[2], a1[2], b0[4], b1[4];
#pragma unroll
  for (int i = 0; i < 2; ++i) a0[i] = load_a_frag(A, lda, row0 + i * 16, 0, lane);
#pragma unroll
  for (int t = 0; t < 4; ++t) b0[t] = load_bt_frag(Wt, ldw, n0 + t * 16, 0, lane);

#pragma unroll
  for (int k0 = 0; k0 + 64 < K; k0 += 64) {
    // prefetch k0+32 into buffer 1
#pragma unroll
    for (int i = 0; i < 2; ++i) a1[i] = load_a_frag(A, lda, row0 + i * 16, k0 + 32, lane);
#pragma unroll
    for (int t = 0; t < 4; ++t) b1[t] = load_bt_frag(Wt, ldw, n0 + t * 16, k0 + 32, lane);
    // compute with buffer 0 (k0)
#pragma unroll
    for (int t = 0; t < 4; ++t) {
      acc[t]     = wmma_bf16(a0[0], b0[t], acc[t]);
      acc[4 + t] = wmma_bf16(a0[1], b0[t], acc[4 + t]);
    }
    // prefetch k0+64 into buffer 0 (always valid: k0 <= K-128)
#pragma unroll
    for (int i = 0; i < 2; ++i) a0[i] = load_a_frag(A, lda, row0 + i * 16, k0 + 64, lane);
#pragma unroll
    for (int t = 0; t < 4; ++t) b0[t] = load_bt_frag(Wt, ldw, n0 + t * 16, k0 + 64, lane);
    // compute with buffer 1 (k0+32)
#pragma unroll
    for (int t = 0; t < 4; ++t) {
      acc[t]     = wmma_bf16(a1[0], b1[t], acc[t]);
      acc[4 + t] = wmma_bf16(a1[1], b1[t], acc[4 + t]);
    }
  }
  // tail: buffer 0 holds K-64; load K-32 and finish
#pragma unroll
  for (int i = 0; i < 2; ++i) a1[i] = load_a_frag(A, lda, row0 + i * 16, K - 32, lane);
#pragma unroll
  for (int t = 0; t < 4; ++t) b1[t] = load_bt_frag(Wt, ldw, n0 + t * 16, K - 32, lane);
#pragma unroll
  for (int t = 0; t < 4; ++t) {
    acc[t]     = wmma_bf16(a0[0], b0[t], acc[t]);
    acc[4 + t] = wmma_bf16(a0[1], b0[t], acc[4 + t]);
  }
#pragma unroll
  for (int t = 0; t < 4; ++t) {
    acc[t]     = wmma_bf16(a1[0], b1[t], acc[t]);
    acc[4 + t] = wmma_bf16(a1[1], b1[t], acc[4 + t]);
  }
}

// ---------------------------------------------------------------------------
// Tensor Data Mover: 2-D tile load Global -> LDS, D# built per ISA 8.3/8.4.
// data_size = 2 bytes; pad_enable inserts 4 dwords (16B) after every 32 dwords
// (128B = one 64-elem row) -> LDS row stride = PAD_LD elements.
// ---------------------------------------------------------------------------
#if __has_builtin(__builtin_amdgcn_tensor_load_to_lds)
#define HAVE_TDM 1
#if __has_include(<hip/amd_detail/amd_gfx1250_TDM.h>)
// amdgpu-toolchain (clang-23): 6-arg form
#define TDM_CALL(g0, g1, z4) \
  __builtin_amdgcn_tensor_load_to_lds((g0), (g1), (z4), (z4), \
                                      (v8i){0,0,0,0,0,0,0,0}, 0)
#else
// ROCm 7.2 (clang-22): 5-arg form
#define TDM_CALL(g0, g1, z4) \
  __builtin_amdgcn_tensor_load_to_lds((g0), (g1), (z4), (z4), 0)
#endif
#else
#define HAVE_TDM 0
#endif

#if HAVE_TDM
__device__ __forceinline__ void tdm_load_2d(unsigned int lds_off, const void* gptr,
                                            unsigned int tile_w, unsigned int tile_h,
                                            unsigned int tensor_w, unsigned int tensor_h,
                                            unsigned long long row_stride) {
  unsigned long long ga = (unsigned long long)gptr;
  v4u g0;
  g0.x = 1u;                                           // count=1, user descriptor
  g0.y = lds_off;                                      // lds_addr (bytes)
  g0.z = (unsigned int)ga;                             // global_addr[31:0]
  g0.w = (unsigned int)((ga >> 32) & 0x1FFFFFFull)     // global_addr[56:32]
         | (2u << 30);                                 // type = 2 ("image")
  v8i g1;
  unsigned int d0 = (1u << 16)                         // data_size = 2 bytes
                  | (1u << 20)                         // pad_enable
                  | (4u << 22)                         // pad_interval: every 32 dwords
                  | (3u << 25);                        // pad_amount: 4 dwords
  g1[0] = (int)d0;
  g1[1] = (int)((tensor_w & 0xFFFFu) << 16);           // tensor_dim0 lo16 (abar=0)
  g1[2] = (int)(((tensor_w >> 16) & 0xFFFFu) | ((tensor_h & 0xFFFFu) << 16));
  g1[3] = (int)(((tensor_h >> 16) & 0xFFFFu) | ((tile_w & 0xFFFFu) << 16));
  g1[4] = (int)(tile_h & 0xFFFFu);                     // tile_dim1 (tile_dim2 = 0)
  g1[5] = (int)(unsigned int)(row_stride & 0xFFFFFFFFull);   // dim0_stride lo32
  g1[6] = (int)(unsigned int)((row_stride >> 32) & 0xFFFFull);
  g1[7] = 0;
  v4i z4 = {0, 0, 0, 0};
  TDM_CALL(g0, g1, z4);
}
#endif

// ---------------------------------------------------------------------------
// Stage 0: fp32 -> bf16 conversion, 4-wide
// ---------------------------------------------------------------------------
__global__ void cvt_f32_to_bf16_x4(const float4* __restrict__ in,
                                   v4bf* __restrict__ out, int n4) {
  int i = blockIdx.x * blockDim.x + threadIdx.x;
  if (i < n4) {
    float4 v = in[i];
    v4bf o;
    o[0] = (bf16_t)v.x; o[1] = (bf16_t)v.y; o[2] = (bf16_t)v.z; o[3] = (bf16_t)v.w;
    out[i] = o;
  }
}

// ---------------------------------------------------------------------------
// Stage 1: QKV projection.  X[8192,768] x Wqkv[2304,768]^T.
// 8 waves of a block share one 64-col weight strip; waves differ in M tile.
// Epilogue: a 32-row tile never crosses a batch boundary (1024%32==0), so
// b/which/h are tile-uniform and every store is one per-lane base pointer
// plus compile-time immediate offsets.
// ---------------------------------------------------------------------------
__global__ __launch_bounds__(256) void qkv_gemm_kernel(
    const bf16_t* __restrict__ Xb, const bf16_t* __restrict__ Wq,
    bf16_t* __restrict__ Qb, bf16_t* __restrict__ Kb, bf16_t* __restrict__ Vt) {
  int wid  = blockIdx.x * 8 + (threadIdx.x >> 5);
  int lane = threadIdx.x & 31;
  int nIdx = wid >> 8;               // 256 m-tiles (8192/32) per 64-col strip
  int mIdx = wid & 255;
  if (nIdx >= QKV_N / 64) return;
  int row0 = mIdx * 32;
  int n0   = nIdx * 64;

  v8f acc[8];
#pragma unroll
  for (int t = 0; t < 8; ++t) acc[t] = (v8f){0.f,0.f,0.f,0.f,0.f,0.f,0.f,0.f};

  gemm_tile_32x64(Xb, DIM, Wq, DIM, row0, n0, DIM, acc, lane);

  int half  = lane >> 4;
  int c     = lane & 15;
  int which = n0 / DIM;              // 0=Q 1=K 2=V (64-wide tile sits in one head)
  int h     = (n0 % DIM) / HDIM;
  int bb    = row0 >> 10;            // batch index, tile-uniform
  int nrow  = row0 & 1023;           // sequence row base
  size_t bh = (size_t)(bb * HEADS + h);

  if (which == 2) {
    // Vt[(bh*64 + d)*SEQ + n], d = t*16+c, n = nrow + rb*16 + r + 8*half
    bf16_t* dst = Vt + (bh * HDIM + c) * SEQ + nrow + 8 * half;
#pragma unroll
    for (int t = 0; t < 4; ++t)
#pragma unroll
      for (int rb = 0; rb < 2; ++rb)
#pragma unroll
        for (int r = 0; r < 8; ++r)
          dst[(size_t)t * 16 * SEQ + rb * 16 + r] = (bf16_t)acc[rb * 4 + t][r];
  } else {
    // (Q|K)[(bh*SEQ + n)*64 + d]
    bf16_t* dst = (which == 0 ? Qb : Kb) +
                  (bh * SEQ + nrow + 8 * half) * HDIM + c;
#pragma unroll
    for (int rb = 0; rb < 2; ++rb)
#pragma unroll
      for (int t = 0; t < 4; ++t)
#pragma unroll
        for (int r = 0; r < 8; ++r)
          dst[(rb * 16 + r) * HDIM + t * 16] = (bf16_t)acc[rb * 4 + t][r];
  }
}

// ---------------------------------------------------------------------------
// Stage 2: flash attention.  4 waves = 64 query rows of one (b,h).
// K,V tiles staged by the Tensor Data Mover into double-buffered LDS; key
// loop unrolled x2 so the buffer index jb&1 is a constant per copy.  One
// barrier/iteration: every ds_read is consumed by a WMMA (forcing its DScnt
// wait) before the barrier, so buf reads complete before the next overwrite.
// ---------------------------------------------------------------------------
__global__ __launch_bounds__(128) void attn_kernel(
    const bf16_t* __restrict__ Qb, const bf16_t* __restrict__ Kb,
    const bf16_t* __restrict__ Vt, bf16_t* __restrict__ Ab) {
  __shared__ __align__(16) bf16_t Kl[2][64 * PAD_LD];  // [key][d]
  __shared__ __align__(16) bf16_t Vl[2][64 * PAD_LD];  // [d][key]
  __shared__ __align__(16) bf16_t Pl[4][16 * PAD_LD];  // per-wave P scratch

  int w    = threadIdx.x >> 5;
  int lane = threadIdx.x & 31;
  int bh   = blockIdx.x >> 4;
  int qb   = blockIdx.x & 15;
  int b    = bh / HEADS;
  int h    = bh % HEADS;

  const bf16_t* Qh = Qb + (size_t)bh * SEQ * HDIM;
  const bf16_t* Kh = Kb + (size_t)bh * SEQ * HDIM;
  const bf16_t* Vh = Vt + (size_t)bh * HDIM * SEQ;
  int row0 = qb * 64 + w * 16;

  // Q fragments: resident in VGPRs for the whole key loop
  v16bf aQ0 = load_a_frag(Qh, HDIM, row0, 0, lane);
  v16bf aQ1 = load_a_frag(Qh, HDIM, row0, 32, lane);

  float mi[8], li[8];
  v8f o[4];
#pragma unroll
  for (int r = 0; r < 8; ++r) { mi[r] = -3.0e38f; li[r] = 0.f; }
#pragma unroll
  for (int t = 0; t < 4; ++t) o[t] = (v8f){0.f,0.f,0.f,0.f,0.f,0.f,0.f,0.f};

  int half = lane >> 4;
  int c    = lane & 15;

#if HAVE_TDM
  if (w == 0) {
    tdm_load_2d((unsigned int)(unsigned long long)(const void*)&Kl[0][0],
                Kh, 64, 64, 64, SEQ, 64);
    tdm_load_2d((unsigned int)(unsigned long long)(const void*)&Vl[0][0],
                Vh, 64, 64, SEQ, 64, SEQ);
  }
#endif

#pragma unroll 2
  for (int jb = 0; jb < SEQ / 64; ++jb) {
    int j0   = jb * 64;
    int buf  = jb & 1;
#if HAVE_TDM
    if (w == 0) __builtin_amdgcn_s_wait_tensorcnt(0);
    __syncthreads();                       // cur tiles visible to all waves
    if (w == 0 && jb + 1 < SEQ / 64) {     // prefetch next block into other buf
      tdm_load_2d((unsigned int)(unsigned long long)(const void*)&Kl[buf ^ 1][0],
                  Kh + (size_t)(j0 + 64) * HDIM, 64, 64, 64, SEQ, 64);
      tdm_load_2d((unsigned int)(unsigned long long)(const void*)&Vl[buf ^ 1][0],
                  Vh + (j0 + 64), 64, 64, SEQ, 64, SEQ);
    }
#else
    __syncthreads();                       // protect previous iteration's reads
    for (int i = threadIdx.x; i < 64 * 8; i += 128) {
      int rr = i >> 3, cc = i & 7;
      *(v8bf*)&Kl[buf][rr * PAD_LD + cc * 8] =
          *(const v8bf*)(Kh + ((size_t)j0 + rr) * HDIM + cc * 8);
      *(v8bf*)&Vl[buf][rr * PAD_LD + cc * 8] =
          *(const v8bf*)(Vh + (size_t)rr * SEQ + j0 + cc * 8);
    }
    __syncthreads();
#endif
    const bf16_t* Kc = &Kl[buf][0];
    const bf16_t* Vc = &Vl[buf][0];

    // ---- S = Q * K^T for 64 keys (B-frags from LDS, conflict-free) ----
    v8f s[4];
#pragma unroll
    for (int t = 0; t < 4; ++t) {
      s[t] = (v8f){0.f,0.f,0.f,0.f,0.f,0.f,0.f,0.f};
      s[t] = wmma_bf16(aQ0, load_bt_frag(Kc, PAD_LD, t * 16, 0, lane), s[t]);
      s[t] = wmma_bf16(aQ1, load_bt_frag(Kc, PAD_LD, t * 16, 32, lane), s[t]);
    }
    // ---- scale + row max ----
    float rm[8];
#pragma unroll
    for (int r = 0; r < 8; ++r) {
#pragma unroll
      for (int t = 0; t < 4; ++t) s[t][r] *= ATTN_SCALE;
      rm[r] = fmaxf(fmaxf(s[0][r], s[1][r]), fmaxf(s[2][r], s[3][r]));
    }
#pragma unroll
    for (int off = 1; off < 16; off <<= 1)
#pragma unroll
      for (int r = 0; r < 8; ++r)
        rm[r] = fmaxf(rm[r], __shfl_xor(rm[r], off, 32));
    // ---- online softmax ----
    float corr[8], rs[8];
#pragma unroll
    for (int r = 0; r < 8; ++r) {
      float mn = fmaxf(mi[r], rm[r]);
      corr[r] = __expf(mi[r] - mn);
      mi[r] = mn;
      rs[r] = 0.f;
    }
#pragma unroll
    for (int t = 0; t < 4; ++t)
#pragma unroll
      for (int r = 0; r < 8; ++r) {
        float p = __expf(s[t][r] - mi[r]);
        s[t][r] = p;
        rs[r] += p;
      }
#pragma unroll
    for (int off = 1; off < 16; off <<= 1)
#pragma unroll
      for (int r = 0; r < 8; ++r)
        rs[r] += __shfl_xor(rs[r], off, 32);
#pragma unroll
    for (int r = 0; r < 8; ++r) li[r] = li[r] * corr[r] + rs[r];
#pragma unroll
    for (int t = 0; t < 4; ++t)
#pragma unroll
      for (int r = 0; r < 8; ++r) o[t][r] *= corr[r];

    // ---- P: C-layout -> A-layout via wave-private LDS (padded rows) ----
#pragma unroll
    for (int t = 0; t < 4; ++t)
#pragma unroll
      for (int r = 0; r < 8; ++r)
        Pl[w][(r + 8 * half) * PAD_LD + t * 16 + c] = (bf16_t)s[t][r];
    v16bf pa0 = load_a_frag(&Pl[w][0], PAD_LD, 0, 0, lane);
    v16bf pa1 = load_a_frag(&Pl[w][0], PAD_LD, 0, 32, lane);

    // ---- O += P * V (B-frags from LDS V tile) ----
#pragma unroll
    for (int t = 0; t < 4; ++t) {
      o[t] = wmma_bf16(pa0, load_bt_frag(Vc, PAD_LD, t * 16, 0, lane), o[t]);
      o[t] = wmma_bf16(pa1, load_bt_frag(Vc, PAD_LD, t * 16, 32, lane), o[t]);
    }
  }

  // ---- normalize, store [B, N, H*D] bf16 (base + immediate offsets) ----
#pragma unroll
  for (int r = 0; r < 8; ++r) li[r] = 1.0f / li[r];
  bf16_t* dst = Ab + ((size_t)(b * SEQ + qb * 64 + w * 16 + 8 * half)) * DIM +
                h * HDIM + c;
#pragma unroll
  for (int t = 0; t < 4; ++t)
#pragma unroll
    for (int r = 0; r < 8; ++r)
      dst[r * DIM + t * 16] = (bf16_t)(o[t][r] * li[r]);
}

// ---------------------------------------------------------------------------
// Stage 3: output projection + bias, fp32 out (base + immediate offsets).
// ---------------------------------------------------------------------------
__global__ __launch_bounds__(256) void proj_gemm_kernel(
    const bf16_t* __restrict__ Ab, const bf16_t* __restrict__ Wp,
    const float* __restrict__ bias, float* __restrict__ out) {
  int wid  = blockIdx.x * 8 + (threadIdx.x >> 5);
  int lane = threadIdx.x & 31;
  int nIdx = wid >> 8;               // 256 m-tiles per 64-col strip
  int mIdx = wid & 255;
  if (nIdx >= DIM / 64) return;
  int row0 = mIdx * 32;
  int n0   = nIdx * 64;

  v8f acc[8];
#pragma unroll
  for (int t = 0; t < 8; ++t) acc[t] = (v8f){0.f,0.f,0.f,0.f,0.f,0.f,0.f,0.f};

  gemm_tile_32x64(Ab, DIM, Wp, DIM, row0, n0, DIM, acc, lane);

  int half = lane >> 4;
  int c    = lane & 15;
  float bv[4];
#pragma unroll
  for (int t = 0; t < 4; ++t) bv[t] = bias[n0 + t * 16 + c];

  float* dst = out + (size_t)(row0 + 8 * half) * DIM + n0 + c;
#pragma unroll
  for (int rb = 0; rb < 2; ++rb)
#pragma unroll
    for (int t = 0; t < 4; ++t)
#pragma unroll
      for (int r = 0; r < 8; ++r)
        dst[(size_t)(rb * 16 + r) * DIM + t * 16] = acc[rb * 4 + t][r] + bv[t];
}

// ---------------------------------------------------------------------------
// Host launcher
// ---------------------------------------------------------------------------
extern "C" void kernel_launch(void* const* d_in, const int* in_sizes, int n_in,
                              void* d_out, int out_size, void* d_ws, size_t ws_size,
                              hipStream_t stream) {
  const float* x      = (const float*)d_in[0];
  const float* w_qkv  = (const float*)d_in[1];
  const float* w_proj = (const float*)d_in[2];
  const float* b_proj = (const float*)d_in[3];
  float* out = (float*)d_out;

  char* ws = (char*)d_ws;
  size_t off = 0;
  bf16_t* Xb  = (bf16_t*)(ws + off); off += (size_t)ROWS * DIM * 2;
  bf16_t* Wqb = (bf16_t*)(ws + off); off += (size_t)QKV_N * DIM * 2;
  bf16_t* Wpb = (bf16_t*)(ws + off); off += (size_t)DIM * DIM * 2;
  bf16_t* Qb  = (bf16_t*)(ws + off); off += (size_t)BATCH * HEADS * SEQ * HDIM * 2;
  bf16_t* Kb  = (bf16_t*)(ws + off); off += (size_t)BATCH * HEADS * SEQ * HDIM * 2;
  bf16_t* Vt  = (bf16_t*)(ws + off); off += (size_t)BATCH * HEADS * HDIM * SEQ * 2;
  bf16_t* Ab  = (bf16_t*)(ws + off); off += (size_t)ROWS * DIM * 2;

  // Stage 0: conversions (all sizes divisible by 4)
  {
    int n0 = ROWS * DIM / 4, n1 = QKV_N * DIM / 4, n2 = DIM * DIM / 4;
    cvt_f32_to_bf16_x4<<<(n0 + 255) / 256, 256, 0, stream>>>((const float4*)x, (v4bf*)Xb, n0);
    cvt_f32_to_bf16_x4<<<(n1 + 255) / 256, 256, 0, stream>>>((const float4*)w_qkv, (v4bf*)Wqb, n1);
    cvt_f32_to_bf16_x4<<<(n2 + 255) / 256, 256, 0, stream>>>((const float4*)w_proj, (v4bf*)Wpb, n2);
  }

  // Stage 1: (8192/32)*(2304/64) = 9216 wave-tiles -> 1152 blocks
  qkv_gemm_kernel<<<1152, 256, 0, stream>>>(Xb, Wqb, Qb, Kb, Vt);

  // Stage 2: B*H*(1024/64) = 1536 blocks of 128 threads
  attn_kernel<<<BATCH * HEADS * (SEQ / 64), 128, 0, stream>>>(Qb, Kb, Vt, Ab);

  // Stage 3: (8192/32)*(768/64) = 3072 wave-tiles -> 384 blocks
  proj_gemm_kernel<<<384, 256, 0, stream>>>(Ab, Wpb, b_proj, out);
}
